// UnsupervisedTriphard_74758200754448
// MI455X (gfx1250) — matrix operand order repaired and checked by reading, not compiled, not run on hardware
//
#include <hip/hip_runtime.h>

// ---------------------------------------------------------------------------
// UnsupervisedTriphard on MI455X (gfx1250, wave32)
//
// Fused  G = X·X^T  (v_wmma_f32_16x16x32_bf16, f32 accum) + streaming
// per-row top-3 mine. Never materializes the 8192x8192 distance matrix.
// Ranking key: sq[n] - 2*dot (sqrt and per-row sq[m] are rank-invariant),
// packed with the column index into one sortable u64 so the top-3 update is
// a branchless min/max chain (v_cndmask, no exec divergence) that
// co-executes with the XDL WMMA pipe. Final d_ap/d_an recomputed in f32.
// ---------------------------------------------------------------------------

typedef __attribute__((ext_vector_type(16))) __bf16 v16bf;
typedef __attribute__((ext_vector_type(8)))  __bf16 v8bf;
typedef __attribute__((ext_vector_type(8)))  float  v8f;
typedef unsigned long long u64;

#define N_PTS   8192
#define DIM     256
#define NT_TOT  (N_PTS / 16)        // 512 column tiles
#define NQ      4                    // N-dimension split for parallelism
#define NT_Q    (NT_TOT / NQ)        // 128 tiles per block
#define ROWS_PB 128                  // rows per block (8 waves x 16)
#define KEY_MAX 0xFFFFFFFFFFFFFFFFull

// float -> order-preserving u32 (total order, -inf..+inf ascending)
__device__ __forceinline__ unsigned int fkey(float f)
{
    unsigned int u = __float_as_uint(f);
    u ^= (unsigned int)(((int)u) >> 31) | 0x80000000u;   // branchless
    return u;
}
__device__ __forceinline__ u64 umin64(u64 a, u64 b) { return a < b ? a : b; }
__device__ __forceinline__ u64 umax64(u64 a, u64 b) { return a < b ? b : a; }

// ---------------------------------------------------------------- prep ------
// f32 -> bf16 convert, row sum-of-squares, zero the output scalar.
__global__ __launch_bounds__(256, 1)
void prep_kernel(const float* __restrict__ X, float* __restrict__ sq,
                 __bf16* __restrict__ Xbf, float* __restrict__ out)
{
    if (blockIdx.x == 0 && threadIdx.x == 0) out[0] = 0.0f;

    const int wave = threadIdx.x >> 5;
    const int lane = threadIdx.x & 31;
    const int row  = blockIdx.x * 8 + wave;

    const float* x = X + row * DIM + lane * 8;
    float4 a = ((const float4*)x)[0];
    float4 b = ((const float4*)x)[1];

    float s = a.x*a.x + a.y*a.y + a.z*a.z + a.w*a.w
            + b.x*b.x + b.y*b.y + b.z*b.z + b.w*b.w;

    v8bf ob;
    ob[0] = (__bf16)a.x; ob[1] = (__bf16)a.y; ob[2] = (__bf16)a.z; ob[3] = (__bf16)a.w;
    ob[4] = (__bf16)b.x; ob[5] = (__bf16)b.y; ob[6] = (__bf16)b.z; ob[7] = (__bf16)b.w;
    *(v8bf*)(Xbf + row * DIM + lane * 8) = ob;

#pragma unroll
    for (int off = 16; off > 0; off >>= 1)
        s += __shfl_xor(s, off, 32);
    if (lane == 0) sq[row] = s;
}

// ---------------------------------------------------------------- mine ------
// Each wave: one 16-row M tile, A panel (16x256 bf16) resident in VGPRs.
// 8 waves share double-buffered 16x256 bf16 B panels through LDS.
// grid = (64 M-strips, 4 N-quarters). Emits 3 u64 candidates/row/quarter.
__global__ __launch_bounds__(256, 1)
void mine_kernel(const __bf16* __restrict__ Xbf, const float* __restrict__ sq,
                 u64* __restrict__ candK)
{
    __shared__ __align__(16) char smem[49152];   // 2x8KB panels, reused for merge
    __bf16* panel = (__bf16*)smem;

    const int tid  = threadIdx.x;
    const int wave = tid >> 5;
    const int lane = tid & 31;
    const int hi   = lane >> 4;     // which 16-lane half
    const int nlo  = lane & 15;
    const int mbase = blockIdx.x * ROWS_PB + wave * 16;
    const int q     = blockIdx.y;
    const int tile0 = q * NT_Q;

    // --- resident A: 8 k-tiles of 16x32 bf16 (contiguous K striping:
    //     lanes 0-15 hold K {0..15}, lanes 16-31 hold K {16..31})
    v16bf areg[8];
    {
        const __bf16* arow = Xbf + (mbase + nlo) * DIM;
#pragma unroll
        for (int kt = 0; kt < 8; ++kt) {
            v8bf c0 = *(const v8bf*)(arow + kt * 32 + hi * 16);
            v8bf c1 = *(const v8bf*)(arow + kt * 32 + hi * 16 + 8);
#pragma unroll
            for (int i = 0; i < 8; ++i) { areg[kt][i] = c0[i]; areg[kt][i + 8] = c1[i]; }
        }
    }

    // per-lane top-3 smallest packed keys per C row r
    u64 k0[8], k1[8], k2[8];
#pragma unroll
    for (int r = 0; r < 8; ++r) { k0[r] = k1[r] = k2[r] = KEY_MAX; }

    // cooperative B-panel staging: thread -> 32 contiguous bytes
    const int prow = tid >> 4;       // 0..15 = column index n within tile
    const int pseg = tid & 15;       // 16-element segment along K
    {
        const __bf16* src = Xbf + (tile0 * 16 + prow) * DIM + pseg * 16;
        int4 r0 = ((const int4*)src)[0];
        int4 r1 = ((const int4*)src)[1];
        __bf16* dst = panel + prow * DIM + pseg * 16;
        ((int4*)dst)[0] = r0; ((int4*)dst)[1] = r1;
    }
    __syncthreads();

    for (int tt = 0; tt < NT_Q; ++tt) {
        const int t = tile0 + tt;

        int4 p0, p1;
        if (tt + 1 < NT_Q) {   // prefetch next panel into regs while computing
            const __bf16* src = Xbf + ((t + 1) * 16 + prow) * DIM + pseg * 16;
            p0 = ((const int4*)src)[0];
            p1 = ((const int4*)src)[1];
        }

        // B tile lane view (contiguous K striping, same as A): lane -> column nlo,
        // K half selected by hi, 16 contiguous bf16 per k-tile.
        const __bf16* bbuf = panel + (tt & 1) * (16 * DIM) + nlo * DIM + hi * 16;

        // preload ALL B fragments, then issue WMMAs back-to-back so one
        // dscnt wait covers 8 matrix ops instead of stalling per-op
        v16bf bregs[8];
#pragma unroll
        for (int kt = 0; kt < 8; ++kt) {
            v8bf b0 = ((const v8bf*)(bbuf + kt * 32))[0];
            v8bf b1 = ((const v8bf*)(bbuf + kt * 32))[1];
#pragma unroll
            for (int i = 0; i < 8; ++i) { bregs[kt][i] = b0[i]; bregs[kt][i + 8] = b1[i]; }
        }

        v8f acc = {0.f, 0.f, 0.f, 0.f, 0.f, 0.f, 0.f, 0.f};
#pragma unroll
        for (int kt = 0; kt < 8; ++kt) {
            acc = __builtin_amdgcn_wmma_f32_16x16x32_bf16(
                      false, areg[kt], false, bregs[kt], (short)0, acc, false, false);
        }

        // mining key: sq[n] - 2*dot packed with column index -> u64;
        // branchless 3-smallest via min/max chain (pure v_cndmask)
        const int   gi  = t * 16 + nlo;
        const float sqn = sq[gi];
#pragma unroll
        for (int r = 0; r < 8; ++r) {
            float d = __builtin_fmaf(-2.0f, acc[r], sqn);
            u64 k = ((u64)fkey(d) << 32) | (unsigned int)gi;
            u64 a0 = umin64(k0[r], k);  u64 b0 = umax64(k0[r], k);  k0[r] = a0;
            u64 a1 = umin64(k1[r], b0); u64 b1 = umax64(k1[r], b0); k1[r] = a1;
            k2[r] = umin64(k2[r], b1);
        }

        if (tt + 1 < NT_Q) {
            __bf16* dst = panel + ((tt + 1) & 1) * (16 * DIM) + prow * DIM + pseg * 16;
            ((int4*)dst)[0] = p0; ((int4*)dst)[1] = p1;
        }
        __syncthreads();
    }

    // --- merge 16 lanes x 3 candidates per row through LDS
    u64* ck = (u64*)smem;    // 128 rows * 48 keys = 49152 B
#pragma unroll
    for (int r = 0; r < 8; ++r) {
        // C/D layout: row = r + 8*hi, column = nlo
        int rowLocal = wave * 16 + r + 8 * hi;
        int base = (rowLocal * 16 + nlo) * 3;
        ck[base + 0] = k0[r];
        ck[base + 1] = k1[r];
        ck[base + 2] = k2[r];
    }
    __syncthreads();

    if (tid < ROWS_PB) {
        const u64* v = ck + tid * 48;
        u64 b0 = KEY_MAX, b1 = KEY_MAX, b2 = KEY_MAX;
        for (int j = 0; j < 48; ++j) {
            u64 k = v[j];
            u64 a0 = umin64(b0, k);  u64 c0 = umax64(b0, k);  b0 = a0;
            u64 a1 = umin64(b1, c0); u64 c1 = umax64(b1, c0); b1 = a1;
            b2 = umin64(b2, c1);
        }
        int rowG = blockIdx.x * ROWS_PB + tid;
        u64* g = candK + rowG * (3 * NQ) + q * 3;
        g[0] = b0; g[1] = b1; g[2] = b2;
    }
}

// --------------------------------------------------------------- select -----
// Merge NQ*3 candidates per row -> third-smallest key; its low 32 bits are
// the hard-negative column index (tie-break = smaller index, like top_k).
__global__ __launch_bounds__(256, 1)
void select_kernel(const u64* __restrict__ candK, int* __restrict__ negIdx)
{
    int row = blockIdx.x * blockDim.x + threadIdx.x;
    if (row >= N_PTS) return;
    const u64* v = candK + row * (3 * NQ);
    u64 b0 = KEY_MAX, b1 = KEY_MAX, b2 = KEY_MAX;
#pragma unroll
    for (int j = 0; j < 3 * NQ; ++j) {
        u64 k = v[j];
        u64 a0 = umin64(b0, k);  u64 c0 = umax64(b0, k);  b0 = a0;
        u64 a1 = umin64(b1, c0); u64 c1 = umax64(b1, c0); b1 = a1;
        b2 = umin64(b2, c1);
    }
    negIdx[row] = (int)(b2 & 0xFFFFFFFFull);
}

// ---------------------------------------------------------------- loss ------
// Exact f32: d_ap, d_an per row, mean of relu(d_ap - d_an + margin).
__global__ __launch_bounds__(256, 1)
void loss_kernel(const float* __restrict__ X, const float* __restrict__ P,
                 const int* __restrict__ negIdx, float* __restrict__ out)
{
    const int wave = threadIdx.x >> 5;
    const int lane = threadIdx.x & 31;
    const int row  = blockIdx.x * 8 + wave;

    const float* x = X + row * DIM;
    const float* p = P + row * DIM;
    const float* n = X + negIdx[row] * DIM;

    float sap = 0.f, san = 0.f;
#pragma unroll
    for (int e = 0; e < 8; ++e) {
        float xv = x[lane + e * 32];
        float dp = xv - p[lane + e * 32] + 1e-6f;
        float dn = xv - n[lane + e * 32] + 1e-6f;
        sap += dp * dp;
        san += dn * dn;
    }
#pragma unroll
    for (int off = 16; off > 0; off >>= 1) {
        sap += __shfl_xor(sap, off, 32);
        san += __shfl_xor(san, off, 32);
    }
    if (lane == 0) {
        float term = fmaxf(sqrtf(sap) - sqrtf(san) + 0.3f, 0.0f);
        atomicAdd(out, term * (1.0f / (float)N_PTS));
    }
}

// ------------------------------------------------------------- launcher -----
extern "C" void kernel_launch(void* const* d_in, const int* in_sizes, int n_in,
                              void* d_out, int out_size, void* d_ws, size_t ws_size,
                              hipStream_t stream)
{
    const float* X = (const float*)d_in[0];
    const float* P = (const float*)d_in[1];
    float* out = (float*)d_out;

    char* ws = (char*)d_ws;
    float*  sq     = (float*)(ws);                                   //  32 KB
    int*    negIdx = (int*)  (ws + (32 << 10));                      //  32 KB
    u64*    candK  = (u64*)  (ws + (64 << 10));                      // 768 KB
    __bf16* Xbf    = (__bf16*)(ws + (64 << 10) + N_PTS * 12 * 8);    //   4 MB

    prep_kernel<<<N_PTS / 8, 256, 0, stream>>>(X, sq, Xbf, out);

    dim3 grid(N_PTS / ROWS_PB, NQ);
    mine_kernel<<<grid, 256, 0, stream>>>(Xbf, sq, candK);

    select_kernel<<<N_PTS / 256, 256, 0, stream>>>(candK, negIdx);

    loss_kernel<<<N_PTS / 8, 256, 0, stream>>>(X, P, negIdx, out);
}
